// Router_5592047420170
// MI455X (gfx1250) — matrix operand order repaired and checked by reading, not compiled, not run on hardware
//
#include <hip/hip_runtime.h>
#include <cstdint>
#include <cstddef>

// ext-vector types matching CDNA5 WMMA operand shapes (wave32)
typedef __attribute__((ext_vector_type(16))) __bf16 v16bf;
typedef __attribute__((ext_vector_type(4)))  __bf16 v4bf;
typedef __attribute__((ext_vector_type(8)))  float  v8f;
typedef __attribute__((ext_vector_type(4)))  float  v4f;   // clang ext-vector (NT-load friendly)

// ---------------------------------------------------------------------------
// Kernel 1: one-shot convert W [E=64, D] f32 -> bf16 into workspace.
// ---------------------------------------------------------------------------
__global__ void w_to_bf16_kernel(const float* __restrict__ w,
                                 unsigned short* __restrict__ o, int n) {
    int i = (blockIdx.x * blockDim.x + threadIdx.x) * 4;
    if (i + 3 < n) {
        v4f f = *(const v4f*)(w + i);
        v4bf v;
        v[0] = (__bf16)f.x; v[1] = (__bf16)f.y;
        v[2] = (__bf16)f.z; v[3] = (__bf16)f.w;
        *(v4bf*)(o + i) = v;
    } else {
        for (; i < n; ++i) {
            __bf16 t = (__bf16)w[i];
            o[i] = *(const unsigned short*)&t;
        }
    }
}

// ---------------------------------------------------------------------------
// Kernel 2: router GEMM (bf16 WMMA) + softmax + top-2, software-pipelined.
// Block = 128 threads = 4 waves; wave w owns tokens [blk*64 + w*16, +16),
// all 64 experts via 4 accumulators of v_wmma_f32_16x16x32_bf16.
// ---------------------------------------------------------------------------
__global__ __launch_bounds__(128) void router_wmma_kernel(
    const float* __restrict__ hs,          // [tokens, D] f32
    const unsigned short* __restrict__ wbf,// [64, D] bf16 (raw u16)
    const float* __restrict__ bias,        // [64] f32
    float* __restrict__ out,               // idx | vals | scores (flat f32)
    int D, long tokens)
{
    __shared__ float sm[64 * 64];          // [token-in-block][expert], 16 KB

    const int tid   = threadIdx.x;
    const int wave  = tid >> 5;
    const int lane  = tid & 31;
    const int lrow  = lane & 15;           // row (A) / column (B,C)
    const int lhalf = lane >> 4;           // lane group 0/1

    const long blockTok = (long)blockIdx.x * 64;
    const long waveTok  = blockTok + (long)wave * 16;

    // A operand: lane holds row M=lrow; lanes 0-15 take K {0..15},
    // lanes 16-31 take K {16..31} of each 32-wide k-step.
    const float* arow = hs + (waveTok + lrow) * (long)D + (lhalf * 16);
    // B operand: lane = expert column, 16 contiguous bf16 K-values per lane.
    const unsigned short* w0 = wbf + (size_t)(lrow +  0) * D + (lhalf * 16);
    const unsigned short* w1 = wbf + (size_t)(lrow + 16) * D + (lhalf * 16);
    const unsigned short* w2 = wbf + (size_t)(lrow + 32) * D + (lhalf * 16);
    const unsigned short* w3 = wbf + (size_t)(lrow + 48) * D + (lhalf * 16);

    v8f acc0 = {}, acc1 = {}, acc2 = {}, acc3 = {};

    // one k-step of compute: cvt f32->bf16 A operand, 4x WMMA
    auto step = [&](const v4f& x0, const v4f& x1, const v4f& x2, const v4f& x3,
                    const v16bf& y0, const v16bf& y1,
                    const v16bf& y2, const v16bf& y3) {
        v16bf a;
        a[0]  = (__bf16)x0.x; a[1]  = (__bf16)x0.y; a[2]  = (__bf16)x0.z; a[3]  = (__bf16)x0.w;
        a[4]  = (__bf16)x1.x; a[5]  = (__bf16)x1.y; a[6]  = (__bf16)x1.z; a[7]  = (__bf16)x1.w;
        a[8]  = (__bf16)x2.x; a[9]  = (__bf16)x2.y; a[10] = (__bf16)x2.z; a[11] = (__bf16)x2.w;
        a[12] = (__bf16)x3.x; a[13] = (__bf16)x3.y; a[14] = (__bf16)x3.z; a[15] = (__bf16)x3.w;
        acc0 = __builtin_amdgcn_wmma_f32_16x16x32_bf16(false, a, false, y0,
                                                       (short)0, acc0, false, false);
        acc1 = __builtin_amdgcn_wmma_f32_16x16x32_bf16(false, a, false, y1,
                                                       (short)0, acc1, false, false);
        acc2 = __builtin_amdgcn_wmma_f32_16x16x32_bf16(false, a, false, y2,
                                                       (short)0, acc2, false, false);
        acc3 = __builtin_amdgcn_wmma_f32_16x16x32_bf16(false, a, false, y3,
                                                       (short)0, acc3, false, false);
    };

    // ---- prologue: loads for k = 0 ----
    v4f A0 = __builtin_nontemporal_load((const v4f*)(arow));
    v4f A1 = __builtin_nontemporal_load((const v4f*)(arow + 4));
    v4f A2 = __builtin_nontemporal_load((const v4f*)(arow + 8));
    v4f A3 = __builtin_nontemporal_load((const v4f*)(arow + 12));
    v16bf B0 = *(const v16bf*)(w0);
    v16bf B1 = *(const v16bf*)(w1);
    v16bf B2 = *(const v16bf*)(w2);
    v16bf B3 = *(const v16bf*)(w3);

    // ---- steady state: issue k0 loads, compute k0-32 ----
    #pragma unroll 2
    for (int k0 = 32; k0 < D; k0 += 32) {
        // deep prefetch of the A stream (3 k-steps ahead) to cover HBM latency
        __builtin_prefetch(arow + k0 + 96, 0, 0);

        v4f nA0 = __builtin_nontemporal_load((const v4f*)(arow + k0));
        v4f nA1 = __builtin_nontemporal_load((const v4f*)(arow + k0 + 4));
        v4f nA2 = __builtin_nontemporal_load((const v4f*)(arow + k0 + 8));
        v4f nA3 = __builtin_nontemporal_load((const v4f*)(arow + k0 + 12));
        v16bf nB0 = *(const v16bf*)(w0 + k0);
        v16bf nB1 = *(const v16bf*)(w1 + k0);
        v16bf nB2 = *(const v16bf*)(w2 + k0);
        v16bf nB3 = *(const v16bf*)(w3 + k0);

        step(A0, A1, A2, A3, B0, B1, B2, B3);

        A0 = nA0; A1 = nA1; A2 = nA2; A3 = nA3;
        B0 = nB0; B1 = nB1; B2 = nB2; B3 = nB3;
    }
    // ---- peeled final k-step ----
    step(A0, A1, A2, A3, B0, B1, B2, B3);

    // ---- C layout: lane%16 = column N (expert), VGPR i = row M (+8 for hi lanes)
    {
        const int tokrow = wave * 16 + lhalf * 8;
        #pragma unroll
        for (int i = 0; i < 8; ++i) {
            sm[(tokrow + i) * 64 +  0 + lrow] = acc0[i];
            sm[(tokrow + i) * 64 + 16 + lrow] = acc1[i];
            sm[(tokrow + i) * 64 + 32 + lrow] = acc2[i];
            sm[(tokrow + i) * 64 + 48 + lrow] = acc3[i];
        }
    }
    __syncthreads();

    // ---- softmax over E=64 + top-2, one thread per token ----
    if (tid < 64) {
        const long g = blockTok + tid;
        float* row = sm + tid * 64;          // this thread owns its row now

        float mx = -3.4e38f;
        #pragma unroll 8
        for (int e = 0; e < 64; ++e) {
            float v = row[e] + bias[e];
            mx = fmaxf(mx, v);
        }
        float sum = 0.f;
        #pragma unroll 8
        for (int e = 0; e < 64; ++e) {
            float p = __expf(row[e] + bias[e] - mx);
            row[e] = p;
            sum += p;
        }
        const float inv = 1.f / sum;

        float* scores = out + tokens * 4 + g * 64;
        float b1v = -1.f, b2v = -1.f;
        int   i1  = 0,    i2  = 0;
        #pragma unroll 8
        for (int e = 0; e < 64; ++e) {
            float s = row[e] * inv;
            scores[e] = s;
            if (s > b1v)      { b2v = b1v; i2 = i1; b1v = s; i1 = e; }
            else if (s > b2v) { b2v = s;   i2 = e; }
        }
        out[2 * g]                  = (float)i1;   // top_k_indices (flattened f32)
        out[2 * g + 1]              = (float)i2;
        out[tokens * 2 + 2 * g]     = b1v;         // top_k_values
        out[tokens * 2 + 2 * g + 1] = b2v;
    }
}

// ---------------------------------------------------------------------------
extern "C" void kernel_launch(void* const* d_in, const int* in_sizes, int n_in,
                              void* d_out, int out_size, void* d_ws, size_t ws_size,
                              hipStream_t stream) {
    const float* hs   = (const float*)d_in[0];   // [2, 8192, D] f32
    const float* W    = (const float*)d_in[1];   // [E, D] f32
    const float* bias = (const float*)d_in[2];   // [E] f32
    float* out = (float*)d_out;

    const int  E      = in_sizes[2];             // 64
    const int  D      = in_sizes[1] / E;         // 4096
    const long tokens = (long)in_sizes[0] / D;   // 16384

    unsigned short* wbf = (unsigned short*)d_ws; // needs E*D*2 = 512 KB scratch

    // 1) W f32 -> bf16 (one-shot, lives in L2 thereafter)
    {
        const int n   = in_sizes[1];
        const int thr = 256;
        const int blk = (int)(((n + 3) / 4 + thr - 1) / thr);
        w_to_bf16_kernel<<<blk, thr, 0, stream>>>(W, wbf, n);
    }

    // 2) fused GEMM + softmax + top-2
    {
        const int blocks = (int)(tokens / 64);   // 256 blocks of 4 waves
        router_wmma_kernel<<<blocks, 128, 0, stream>>>(hs, wbf, bias, out,
                                                       D, tokens);
    }
}
